// Att2in2Core_9483287790098
// MI455X (gfx1250) — matrix operand order repaired and checked by reading, not compiled, not run on hardware
//
#include <hip/hip_runtime.h>
#include <cstdint>
#include <cstddef>

// Problem constants (from reference): B=256, S=49, H=2048, V_SZ=10000
#define B_SZ 256
#define S_SZ 49
#define H_SZ 2048
#define K_SZ 2048   // every GEMM in this model has K = H = 2048

typedef __attribute__((ext_vector_type(16))) __bf16 v16bf;
typedef __attribute__((ext_vector_type(8)))  float  v8f;
typedef __attribute__((ext_vector_type(4)))  unsigned int v4u;
typedef __attribute__((ext_vector_type(8)))  int    v8i;
typedef __attribute__((ext_vector_type(4)))  int    v4i;

#if __has_builtin(__builtin_amdgcn_tensor_load_to_lds)
#define HAS_TDM 1
#else
#define HAS_TDM 0
#endif

__device__ __forceinline__ unsigned short f2bf(float x) {
  unsigned int u = __float_as_uint(x);
  u += 0x7FFFu + ((u >> 16) & 1u);          // round-to-nearest-even
  return (unsigned short)(u >> 16);
}
__device__ __forceinline__ float bf2f(unsigned short h) {
  return __uint_as_float(((unsigned int)h) << 16);
}

// ---------------------------------------------------------------------------
// f32 -> bf16 streaming conversion (V and all weight matrices, once)
// ---------------------------------------------------------------------------
__global__ __launch_bounds__(256) void convert_f32_bf16(
    const float* __restrict__ in, unsigned short* __restrict__ outp, int n4) {
  int idx = blockIdx.x * 256 + threadIdx.x;
  if (idx >= n4) return;
  float4 v = ((const float4*)in)[idx];
  uint2 p;
  p.x = (unsigned)f2bf(v.x) | ((unsigned)f2bf(v.y) << 16);
  p.y = (unsigned)f2bf(v.z) | ((unsigned)f2bf(v.w) << 16);
  ((uint2*)outp)[idx] = p;
}

__global__ __launch_bounds__(256) void zero_f32(float* __restrict__ p, int n) {
  int i = blockIdx.x * 256 + threadIdx.x;
  if (i < n) p[i] = 0.f;
}

// ---------------------------------------------------------------------------
// xt = relu(emb[captions[:,time]]) -> bf16 ; hprev = (time==0 ? 0 : state[0,0]) -> bf16
// ---------------------------------------------------------------------------
__global__ __launch_bounds__(256) void prep_xt_h(
    const int* __restrict__ caps, const int* __restrict__ timep,
    const float* __restrict__ emb, const float* __restrict__ state,
    unsigned short* __restrict__ xt, unsigned short* __restrict__ hprev) {
  int idx = blockIdx.x * 256 + threadIdx.x;          // over B*H
  int b = idx >> 11, h = idx & (H_SZ - 1);
  int t = timep[0];
  int it = caps[b * 16 + t];
  float e = emb[(size_t)it * H_SZ + h];
  xt[idx] = f2bf(e > 0.f ? e : 0.f);
  float hp = (t == 0) ? 0.f : state[idx];            // state[0,0,b,h]
  hprev[idx] = f2bf(hp);
}

// ---------------------------------------------------------------------------
// TDM: issue a 2D tile load (tile_rows x 32 bf16 elems) into LDS.
// D# per CDNA5 ISA ch.8: group0 = {count/lds_addr/global_addr/type},
// group1 = {data_size=2B, pad_enable, pad_interval=16DW, pad_amount=4DW,
//           tensor_dim0=K, tensor_dim1=rows, tile_dim0=32, tile_dim1=tile_rows,
//           tensor_dim0_stride=K}. Padding gives an 80B LDS row stride
// (= 40 ushorts) -> bank-conflict-free fragment reads, matching As/Bs below.
// ---------------------------------------------------------------------------
__device__ __forceinline__ void tdm_tile_load(
    unsigned lds_off, const unsigned short* gptr, int rowlen_elems,
    int tensor_rows, int tile_rows) {
#if HAS_TDM
  unsigned long long ga = (unsigned long long)(uintptr_t)gptr;
  v4u g0;
  g0[0] = 1u;                                          // count=1, user D#
  g0[1] = lds_off;                                     // LDS byte address
  g0[2] = (unsigned)ga;                                // global_addr[31:0]
  g0[3] = (unsigned)((ga >> 32) & 0x01FFFFFFu) | (2u << 30);  // addr[56:32], type=2
  unsigned K = (unsigned)rowlen_elems, R = (unsigned)tensor_rows;
  v8i g1;
  g1[0] = (int)((1u << 16) | (1u << 20) | (3u << 22) | (3u << 25));
  g1[1] = (int)((K & 0xFFFFu) << 16);                  // tensor_dim0[15:0]
  g1[2] = (int)(((K >> 16) & 0xFFFFu) | ((R & 0xFFFFu) << 16));
  g1[3] = (int)(((R >> 16) & 0xFFFFu) | (32u << 16));  // tile_dim0 = 32
  g1[4] = (int)((unsigned)tile_rows & 0xFFFFu);        // tile_dim1, tile_dim2=0
  g1[5] = (int)K;                                      // tensor_dim0_stride[31:0]
  g1[6] = 0;
  g1[7] = 0;
  v4i z4 = {};
#if defined(__clang_major__) && (__clang_major__ >= 23)
  v8i z8 = {};
  __builtin_amdgcn_tensor_load_to_lds(g0, g1, z4, z4, z8, 0);
#else
  __builtin_amdgcn_tensor_load_to_lds(g0, g1, z4, z4, 0);
#endif
#else
  (void)lds_off; (void)gptr; (void)rowlen_elems; (void)tensor_rows; (void)tile_rows;
#endif
}

// ---------------------------------------------------------------------------
// Tiled bf16 WMMA GEMM:  C[M,N] = A[M,K](bf16) @ W[N,K]^T(bf16)
// Block tile 128x128, BK=32, 256 threads = 8 waves in 4(m) x 2(n) grid,
// per-wave 32x64 = 2x4 fragments of v_wmma_f32_16x16x32_bf16.
// Double-buffered LDS filled by the Tensor Data Mover (wave 0 issues,
// s_wait_tensorcnt + workgroup barrier per K-step). The K loop is kept
// NOT-unrolled (runtime buffer offset) so accumulators stay pinned in one
// register range -> no v_mov_b64 accumulator shuffling between iterations.
// MODE 0: out_bf16 = bf16(relu(acc + bias))            (att_feats)
// MODE 1: out_f32  = acc + bias                        (i2h sums / att_h / a2c)
// MODE 2: out_f32 += acc + bias                        (h2h into sums)
// MODE 3: scores[row] += sum_n tanh(acc+bias+extra)*alpha   (fused p_att path)
// ---------------------------------------------------------------------------
union FragAB { v16bf v; uint4 q[2]; };

template <int MODE>
__global__ __launch_bounds__(256) void gemm_xwT(
    const unsigned short* __restrict__ A, const unsigned short* __restrict__ Wb,
    const float* __restrict__ bias, int M, int N,
    float* __restrict__ out_f32, unsigned short* __restrict__ out_bf16,
    const float* __restrict__ extra,   // att_h [B][H]  (MODE 3)
    const float* __restrict__ alpha,   // [N]           (MODE 3)
    float* __restrict__ scores)        // [M]           (MODE 3)
{
  const int K = K_SZ;
  // 40 ushorts/row = 80B stride: 16B aligned, bank-conflict-free
  __shared__ __align__(16) unsigned short As[2][128][40];
  __shared__ __align__(16) unsigned short Bs[2][128][40];

  const int tid  = threadIdx.x;
  const int lane = tid & 31;
  const int wid  = tid >> 5;
  const int wm   = wid & 3;          // 0..3  -> 32-row wave tile
  const int wn   = wid >> 2;         // 0..1  -> 64-col wave tile
  const int half = lane >> 4;        // wave32 half (ISA fragment layout)
  const int l16  = lane & 15;

  const int m0 = blockIdx.y * 128;
  const int n0 = blockIdx.x * 128;
  const int nk = K / 32;
  const int BUFE = 128 * 40;                      // ushorts per buffer

  const unsigned short* AsBase = &As[0][0][0];
  const unsigned short* BsBase = &Bs[0][0][0];
  const unsigned asOff = (unsigned)(uintptr_t)AsBase;   // LDS byte offsets
  const unsigned bsOff = (unsigned)(uintptr_t)BsBase;

  // precomputed per-wave fragment row offsets (in ushorts, within a buffer)
  const int aRow0 = (wm * 32 + l16) * 40;               // fm=0 row
  const int nRow0 = (wn * 64 + l16) * 40;               // fn=0 row
  const int aK0   = half * 16;                          // A chunk-0 k offset
  const int bK0   = half * 16;                          // B chunk-0 k offset

  // prologue: stage tile kt=0 into buffer 0
#if HAS_TDM
  if (wid == 0) {
    tdm_tile_load(asOff, A + (size_t)m0 * K, K, M, 128);
    tdm_tile_load(bsOff, Wb + (size_t)n0 * K, K, N, 128);
    __builtin_amdgcn_s_wait_tensorcnt(0);
  }
#else
#pragma unroll
  for (int j = 0; j < 2; ++j) {
    int c = tid + 256 * j; int r = c >> 2; int col = (c & 3) * 8;
    *(uint4*)&As[0][r][col] = *(const uint4*)(A + (size_t)(m0 + r) * K + col);
    *(uint4*)&Bs[0][r][col] = *(const uint4*)(Wb + (size_t)(n0 + r) * K + col);
  }
#endif
  __syncthreads();

  v8f acc[2][4];
#pragma unroll
  for (int i = 0; i < 2; ++i)
#pragma unroll
    for (int j = 0; j < 4; ++j) acc[i][j] = {};

#pragma unroll 1
  for (int kt = 0; kt < nk; ++kt) {
    const int bofs = (kt & 1) * BUFE;          // runtime double-buffer select
    const int nofs = BUFE - bofs;              // the other buffer
    const bool has_next = (kt + 1) < nk;

    // stage next tile into the other buffer while computing this one
#if HAS_TDM
    if (wid == 0 && has_next) {
      const int kb = (kt + 1) * 32;
      tdm_tile_load(asOff + (unsigned)(nofs * 2), A + (size_t)m0 * K + kb, K, M, 128);
      tdm_tile_load(bsOff + (unsigned)(nofs * 2), Wb + (size_t)n0 * K + kb, K, N, 128);
    }
#else
    if (has_next) {
      const int kb = (kt + 1) * 32;
#pragma unroll
      for (int j = 0; j < 2; ++j) {
        int c = tid + 256 * j; int r = c >> 2; int col = (c & 3) * 8;
        *(uint4*)(const_cast<unsigned short*>(AsBase) + nofs + r * 40 + col) =
            *(const uint4*)(A + (size_t)(m0 + r) * K + kb + col);
        *(uint4*)(const_cast<unsigned short*>(BsBase) + nofs + r * 40 + col) =
            *(const uint4*)(Wb + (size_t)(n0 + r) * K + kb + col);
      }
    }
#endif

    // fragment loads from LDS (contiguous k = 16*half + s, same in A and B) + 8 WMMAs
    const unsigned short* Asb = AsBase + bofs;
    const unsigned short* Bsb = BsBase + bofs;
    FragAB afr[2], bfr[4];
#pragma unroll
    for (int fm = 0; fm < 2; ++fm) {
      const unsigned short* p = Asb + aRow0 + fm * (16 * 40);
      afr[fm].q[0] = *(const uint4*)(p + aK0);           // K = h*16 .. +7
      afr[fm].q[1] = *(const uint4*)(p + aK0 + 8);       // K = h*16+8 .. +15
    }
#pragma unroll
    for (int fn = 0; fn < 4; ++fn) {
      const unsigned short* p = Bsb + nRow0 + fn * (16 * 40);
      bfr[fn].q[0] = *(const uint4*)(p + bK0);           // K = h*16 .. +7
      bfr[fn].q[1] = *(const uint4*)(p + bK0 + 8);       // K = h*16+8 .. +15
    }
#pragma unroll
    for (int fm = 0; fm < 2; ++fm)
#pragma unroll
      for (int fn = 0; fn < 4; ++fn)
        acc[fm][fn] = __builtin_amdgcn_wmma_f32_16x16x32_bf16(
            false, afr[fm].v, false, bfr[fn].v, (short)0, acc[fm][fn], false, false);

#if HAS_TDM
    if (wid == 0 && has_next) __builtin_amdgcn_s_wait_tensorcnt(0);
#endif
    __syncthreads();
  }

  // ---------------- epilogues ----------------
  if (MODE == 3) {
    // fused: scores[row] += sum_cols tanh(acc + ctx_b + att_h[b]) * alpha
#pragma unroll
    for (int fm = 0; fm < 2; ++fm) {
      const int mbase = m0 + wm * 32 + fm * 16 + half * 8;
      float rsum[8];
#pragma unroll
      for (int r = 0; r < 8; ++r) rsum[r] = 0.f;
#pragma unroll
      for (int fn = 0; fn < 4; ++fn) {
        const int col = n0 + wn * 64 + fn * 16 + l16;
        const float bcol = bias[col];
        const float acol = alpha[col];
#pragma unroll
        for (int r = 0; r < 8; ++r) {
          const int row = mbase + r;
          const int bidx = row / S_SZ;
          const float v = acc[fm][fn][r] + bcol + extra[(size_t)bidx * H_SZ + col];
          rsum[r] += tanhf(v) * acol;
        }
      }
      // reduce across the 16 lanes of this half (wave32 shuffles)
#pragma unroll
      for (int off = 1; off < 16; off <<= 1)
#pragma unroll
        for (int r = 0; r < 8; ++r) rsum[r] += __shfl_xor(rsum[r], off, 32);
      if (l16 == 0) {
#pragma unroll
        for (int r = 0; r < 8; ++r) atomicAdd(&scores[mbase + r], rsum[r]);
      }
    }
  } else {
#pragma unroll
    for (int fm = 0; fm < 2; ++fm) {
#pragma unroll
      for (int fn = 0; fn < 4; ++fn) {
        const int col = n0 + wn * 64 + fn * 16 + l16;
        const float bcol = bias[col];
#pragma unroll
        for (int r = 0; r < 8; ++r) {
          const int row = m0 + wm * 32 + fm * 16 + half * 8 + r;
          float v = acc[fm][fn][r] + bcol;
          if (MODE == 0) {
            v = v > 0.f ? v : 0.f;
            out_bf16[(size_t)row * N + col] = f2bf(v);
          } else if (MODE == 1) {
            out_f32[(size_t)row * N + col] = v;
          } else {  // MODE == 2
            out_f32[(size_t)row * N + col] += v;
          }
        }
      }
    }
  }
}

// ---------------------------------------------------------------------------
// softmax over S=49 per batch row (single wave, shuffle reductions)
// ---------------------------------------------------------------------------
__global__ __launch_bounds__(32) void softmax49(
    const float* __restrict__ scores, float* __restrict__ weight) {
  const int b = blockIdx.x, lane = threadIdx.x;
  float s0 = (lane < S_SZ) ? scores[b * S_SZ + lane] : -3.0e38f;
  float s1 = (lane + 32 < S_SZ) ? scores[b * S_SZ + lane + 32] : -3.0e38f;
  float m = fmaxf(s0, s1);
#pragma unroll
  for (int off = 16; off > 0; off >>= 1) m = fmaxf(m, __shfl_xor(m, off, 32));
  float e0 = (lane < S_SZ) ? __expf(s0 - m) : 0.f;
  float e1 = (lane + 32 < S_SZ) ? __expf(s1 - m) : 0.f;
  float sum = e0 + e1;
#pragma unroll
  for (int off = 16; off > 0; off >>= 1) sum += __shfl_xor(sum, off, 32);
  const float inv = 1.0f / sum;
  if (lane < S_SZ) weight[b * S_SZ + lane] = e0 * inv;
  if (lane + 32 < S_SZ) weight[b * S_SZ + lane + 32] = e1 * inv;
}

// ---------------------------------------------------------------------------
// att_res[b,h] = sum_s weight[b,s] * att_feats[b,s,h]   (bf16 out, for a2c GEMM)
// ---------------------------------------------------------------------------
__global__ __launch_bounds__(256) void att_res_k(
    const float* __restrict__ w, const unsigned short* __restrict__ af,
    unsigned short* __restrict__ attres) {
  int idx = blockIdx.x * 256 + threadIdx.x;     // over B*H
  int b = idx >> 11, h = idx & (H_SZ - 1);
  const unsigned short* p = af + (size_t)b * S_SZ * H_SZ + h;
  const float* wb = w + b * S_SZ;
  float accv = 0.f;
#pragma unroll 7
  for (int s = 0; s < S_SZ; ++s) accv += wb[s] * bf2f(p[(size_t)s * H_SZ]);
  attres[idx] = f2bf(accv);
}

// ---------------------------------------------------------------------------
// LSTM cell tail: gates, in_tr max-merge, c/h update, write all 3 outputs
// ---------------------------------------------------------------------------
__global__ __launch_bounds__(256) void lstm_final(
    const float* __restrict__ sums, const float* __restrict__ a2c,
    const float* __restrict__ state, const int* __restrict__ timep,
    float* __restrict__ out) {
  int idx = blockIdx.x * 256 + threadIdx.x;     // over B*H
  int b = idx >> 11, h = idx & (H_SZ - 1);
  const float* sb = sums + (size_t)b * 5 * H_SZ;
  float ig = 1.f / (1.f + __expf(-sb[h]));
  float fg = 1.f / (1.f + __expf(-sb[H_SZ + h]));
  float og = 1.f / (1.f + __expf(-sb[2 * H_SZ + h]));
  float t1 = sb[3 * H_SZ + h] + a2c[(size_t)b * 2 * H_SZ + h];
  float t2 = sb[4 * H_SZ + h] + a2c[(size_t)b * 2 * H_SZ + H_SZ + h];
  float intr = fmaxf(t1, t2);
  float cprev = (timep[0] == 0) ? 0.f : state[(size_t)B_SZ * H_SZ + idx];  // state[1,0,b,h]
  float nc = fg * cprev + ig * intr;
  float nh = og * tanhf(nc);
  out[idx] = nh;                                   // output (B,1,H)
  out[(size_t)B_SZ * H_SZ + idx] = nh;             // next_h (1,B,H)
  out[2 * (size_t)B_SZ * H_SZ + idx] = nc;         // next_c (1,B,H)
}

// ---------------------------------------------------------------------------
extern "C" void kernel_launch(void* const* d_in, const int* in_sizes, int n_in,
                              void* d_out, int out_size, void* d_ws, size_t ws_size,
                              hipStream_t stream) {
  (void)in_sizes; (void)n_in; (void)out_size; (void)ws_size;
  const float* V       = (const float*)d_in[0];
  // d_in[1] v_g, d_in[5] fc_W, d_in[6] fc_b: dead (fc_feats unused by outputs)
  const float* state   = (const float*)d_in[2];
  const int*   caps    = (const int*)d_in[3];
  const int*   timep   = (const int*)d_in[4];
  const float* att_W   = (const float*)d_in[7];
  const float* att_b   = (const float*)d_in[8];
  const float* emb     = (const float*)d_in[9];
  const float* ctx_W   = (const float*)d_in[10];
  const float* ctx_b   = (const float*)d_in[11];
  const float* a2c_W   = (const float*)d_in[12];
  const float* a2c_b   = (const float*)d_in[13];
  const float* i2h_W   = (const float*)d_in[14];
  const float* i2h_b   = (const float*)d_in[15];
  const float* h2h_W   = (const float*)d_in[16];
  const float* h2h_b   = (const float*)d_in[17];
  const float* h2att_W = (const float*)d_in[18];
  const float* h2att_b = (const float*)d_in[19];
  const float* alpha_W = (const float*)d_in[20];
  // alpha_b is softmax-invariant -> skipped
  float* out = (float*)d_out;

  // workspace carve-out (256B aligned)
  char* ws = (char*)d_ws;
  size_t off = 0;
  auto carve = [&](size_t bytes) -> void* {
    off = (off + 255) & ~(size_t)255;
    void* p = ws + off;
    off += bytes;
    return p;
  };
  const size_t MS = (size_t)B_SZ * S_SZ;          // 12544 rows
  const size_t HH = (size_t)H_SZ * H_SZ;
  unsigned short* V_bf     = (unsigned short*)carve(MS * H_SZ * 2);
  unsigned short* attf     = (unsigned short*)carve(MS * H_SZ * 2);
  unsigned short* xt       = (unsigned short*)carve((size_t)B_SZ * H_SZ * 2);
  unsigned short* hprev    = (unsigned short*)carve((size_t)B_SZ * H_SZ * 2);
  unsigned short* attres   = (unsigned short*)carve((size_t)B_SZ * H_SZ * 2);
  unsigned short* attW_bf  = (unsigned short*)carve(HH * 2);
  unsigned short* ctxW_bf  = (unsigned short*)carve(HH * 2);
  unsigned short* h2attW_bf= (unsigned short*)carve(HH * 2);
  unsigned short* i2hW_bf  = (unsigned short*)carve(5 * HH * 2);
  unsigned short* h2hW_bf  = (unsigned short*)carve(5 * HH * 2);
  unsigned short* a2cW_bf  = (unsigned short*)carve(2 * HH * 2);
  float* att_h  = (float*)carve((size_t)B_SZ * H_SZ * 4);
  float* sums   = (float*)carve((size_t)B_SZ * 5 * H_SZ * 4);
  float* a2c    = (float*)carve((size_t)B_SZ * 2 * H_SZ * 4);
  float* scores = (float*)carve(MS * 4);
  float* weight = (float*)carve(MS * 4);

  // one-shot bf16 conversions (streaming, HBM-bound)
  const int n4V = (int)(MS * H_SZ / 4);
  convert_f32_bf16<<<n4V / 256, 256, 0, stream>>>(V, V_bf, n4V);
  const int n4H = (int)(HH / 4);
  convert_f32_bf16<<<n4H / 256, 256, 0, stream>>>(att_W, attW_bf, n4H);
  convert_f32_bf16<<<n4H / 256, 256, 0, stream>>>(ctx_W, ctxW_bf, n4H);
  convert_f32_bf16<<<n4H / 256, 256, 0, stream>>>(h2att_W, h2attW_bf, n4H);
  convert_f32_bf16<<<5 * n4H / 256, 256, 0, stream>>>(i2h_W, i2hW_bf, 5 * n4H);
  convert_f32_bf16<<<5 * n4H / 256, 256, 0, stream>>>(h2h_W, h2hW_bf, 5 * n4H);
  convert_f32_bf16<<<2 * n4H / 256, 256, 0, stream>>>(a2c_W, a2cW_bf, 2 * n4H);

  prep_xt_h<<<(B_SZ * H_SZ) / 256, 256, 0, stream>>>(caps, timep, emb, state, xt, hprev);
  zero_f32<<<(int)(MS / 256), 256, 0, stream>>>(scores, (int)MS);

  // att_feats = relu(V @ att_W^T + att_b)  -> bf16        [12544 x 2048]
  gemm_xwT<0><<<dim3(H_SZ / 128, (int)(MS / 128)), 256, 0, stream>>>(
      V_bf, attW_bf, att_b, (int)MS, H_SZ, nullptr, attf, nullptr, nullptr, nullptr);

  // att_h = hprev @ h2att_W^T + h2att_b                   [256 x 2048]
  gemm_xwT<1><<<dim3(H_SZ / 128, B_SZ / 128), 256, 0, stream>>>(
      hprev, h2attW_bf, h2att_b, B_SZ, H_SZ, att_h, nullptr, nullptr, nullptr, nullptr);

  // fused: scores += sum_h tanh(att_feats@ctx_W^T + ctx_b + att_h)*alpha_W
  gemm_xwT<3><<<dim3(H_SZ / 128, (int)(MS / 128)), 256, 0, stream>>>(
      attf, ctxW_bf, ctx_b, (int)MS, H_SZ, nullptr, nullptr, att_h, alpha_W, scores);

  softmax49<<<B_SZ, 32, 0, stream>>>(scores, weight);
  att_res_k<<<(B_SZ * H_SZ) / 256, 256, 0, stream>>>(weight, attf, attres);

  // sums = xt @ i2h_W^T + i2h_b                           [256 x 10240]
  gemm_xwT<1><<<dim3((5 * H_SZ) / 128, B_SZ / 128), 256, 0, stream>>>(
      xt, i2hW_bf, i2h_b, B_SZ, 5 * H_SZ, sums, nullptr, nullptr, nullptr, nullptr);
  // sums += hprev @ h2h_W^T + h2h_b
  gemm_xwT<2><<<dim3((5 * H_SZ) / 128, B_SZ / 128), 256, 0, stream>>>(
      hprev, h2hW_bf, h2h_b, B_SZ, 5 * H_SZ, sums, nullptr, nullptr, nullptr, nullptr);
  // a2c = att_res @ a2c_W^T + a2c_b                       [256 x 4096]
  gemm_xwT<1><<<dim3((2 * H_SZ) / 128, B_SZ / 128), 256, 0, stream>>>(
      attres, a2cW_bf, a2c_b, B_SZ, 2 * H_SZ, a2c, nullptr, nullptr, nullptr, nullptr);

  lstm_final<<<(B_SZ * H_SZ) / 256, 256, 0, stream>>>(sums, a2c, state, timep, out);
}
